// GNOTBlock_37915971289752
// MI455X (gfx1250) — matrix operand order repaired and checked by reading, not compiled, not run on hardware
//
#include <hip/hip_runtime.h>
#include <hip/hip_bf16.h>
#include <math.h>

// Problem constants (match reference)
#define BB_   4
#define NN_   2048
#define MM_   1024
#define CC    256
#define HH    8
#define DDIM  32
#define HIDN  1024
#define EE    8
#define GHN   512
#define LN_EPS 1e-5f

typedef __attribute__((ext_vector_type(16))) _Float16 v16h;
typedef __attribute__((ext_vector_type(8)))  float    v8f;
typedef _Float16 h16;

__device__ __forceinline__ float gelu_exact(float x) {
  return 0.5f * x * (1.0f + erff(x * 0.70710678118654752f));
}

// A-fragment (16x32 f16, row-major in LDS, leading dim ld):
// lane 0-15: row=lane, K = 0..7 and 16..23 ; lane 16-31: row=lane-16, K = 8..15 and 24..31
// Two contiguous 8-half runs -> lowers to 2x ds_load_b128.
__device__ __forceinline__ v16h load_a_frag(const h16* base, int ld) {
  int lane = threadIdx.x & 31;
  int row  = lane & 15;
  int kb   = (lane >> 4) << 3;
  const h16* p = base + row * ld + kb;
  v16h a;
#pragma unroll
  for (int j = 0; j < 8; ++j) a[j] = p[j];
#pragma unroll
  for (int j = 0; j < 8; ++j) a[8 + j] = p[16 + j];
  return a;
}

// B-fragment from LDS stored in fragment-native layout [col][32 k]:
// lanes 0-15 hold K=0..7,16..23 of col=lane; lanes 16-31 hold K=8..15,24..31.
// Two contiguous 8-half runs -> lowers to 2x ds_load_b128.
__device__ __forceinline__ v16h load_b_fragT(const h16* tileBase) {
  int lane = threadIdx.x & 31;
  int col  = lane & 15;
  int kb   = (lane >> 4) << 3;
  const h16* p = tileBase + col * 32 + kb;
  v16h b;
#pragma unroll
  for (int j = 0; j < 8; ++j) b[j] = p[j];
#pragma unroll
  for (int j = 0; j < 8; ++j) b[8 + j] = p[16 + j];
  return b;
}

__device__ __forceinline__ v8f wmma16(v16h a, v16h b, v8f c) {
  return __builtin_amdgcn_wmma_f32_16x16x32_f16(false, a, false, b, (short)0, c,
                                                false, false);
}

// ---------------- elementwise f32 -> f16 ----------------
__global__ void cvt_f32_f16(const float* __restrict__ src, h16* __restrict__ dst, int n) {
  int i = blockIdx.x * blockDim.x + threadIdx.x;
  if (i < n) dst[i] = (h16)src[i];
}

// ---------------- LayerNorm over C=256, output f16 ----------------
__global__ void ln_kernel(const float* __restrict__ x, const float* __restrict__ g,
                          const float* __restrict__ b, h16* __restrict__ out) {
  __shared__ float s[CC];
  int row = blockIdx.x, t = threadIdx.x;
  float v = x[row * CC + t];
  s[t] = v; __syncthreads();
  for (int off = 128; off > 0; off >>= 1) { if (t < off) s[t] += s[t + off]; __syncthreads(); }
  float mu = s[0] * (1.0f / CC);
  __syncthreads();
  float d = v - mu;
  s[t] = d * d; __syncthreads();
  for (int off = 128; off > 0; off >>= 1) { if (t < off) s[t] += s[t + off]; __syncthreads(); }
  float var = s[0] * (1.0f / CC);
  out[row * CC + t] = (h16)(d * rsqrtf(var + LN_EPS) * g[t] + b[t]);
}

// ---------------- generic WMMA GEMM: out = A(MxK,f16) @ B(KxN,f16) + bias ----------------
// Block tile 64x128, 8 waves; each wave owns a 16x64 strip -> 4 WMMAs per k-step.
// epi: 0 -> f32 out = acc+bias
//      1 -> f16 out = gelu(acc+bias)
//      2 -> f32 out = acc+bias+resid
//      3 -> f16 out = acc+bias
__global__ void __launch_bounds__(256)
gemm_kernel(const h16* __restrict__ A, const h16* __restrict__ Bm,
            const float* __restrict__ bias, const float* __restrict__ resid,
            float* __restrict__ outF, h16* __restrict__ outH,
            int M, int N, int K, int epi) {
  __shared__ h16 As[64 * 32];     // 4KB  row-major [row][k]
  __shared__ h16 BsT[128 * 32];   // 8KB  fragment layout [col][k]
  int t = threadIdx.x;
  int rows0 = blockIdx.y * 64;
  int cols0 = blockIdx.x * 128;
  int w = t >> 5, lane = t & 31;
  int rt = w & 3;          // 4 row tiles of 16
  int cq = w >> 2;         // 2 col quads of 64
  // staging coords (contiguous elements per thread)
  int arow = t >> 2, acol0 = (t & 3) * 8;          // A: 64 rows x 32 k, 8/thread
  int bk   = t >> 3, bcol0 = (t & 7) * 16;         // B: 32 k x 128 cols, 16/thread
  v8f acc[4];
  v8f zero = {};
#pragma unroll
  for (int i = 0; i < 4; ++i) acc[i] = zero;
  for (int kk = 0; kk < K; kk += 32) {
    if (kk + 32 < K) {  // prefetch next k-chunk (global_prefetch_b8)
      __builtin_prefetch(&A[(size_t)(rows0 + arow) * K + kk + 32 + acol0], 0, 1);
      __builtin_prefetch(&Bm[(size_t)(kk + 32 + bk) * N + cols0 + bcol0], 0, 1);
    }
#pragma unroll
    for (int i = 0; i < 8; ++i)
      As[arow * 32 + acol0 + i] = A[(size_t)(rows0 + arow) * K + kk + acol0 + i];
#pragma unroll
    for (int i = 0; i < 16; ++i)
      BsT[(bcol0 + i) * 32 + bk] = Bm[(size_t)(kk + bk) * N + cols0 + bcol0 + i];
    __syncthreads();
    v16h a = load_a_frag(As + rt * 16 * 32, 32);
#pragma unroll
    for (int i = 0; i < 4; ++i) {
      v16h b = load_b_fragT(BsT + (cq * 64 + i * 16) * 32);
      acc[i] = wmma16(a, b, acc[i]);
    }
    __syncthreads();
  }
  int rbase = rows0 + rt * 16 + ((lane >> 4) << 3);
#pragma unroll
  for (int i = 0; i < 4; ++i) {
    int col = cols0 + cq * 64 + i * 16 + (lane & 15);
    float bi = bias[col];
#pragma unroll
    for (int r = 0; r < 8; ++r) {
      int row = rbase + r;
      size_t ix = (size_t)row * N + col;
      float v = acc[i][r] + bi;
      if (epi == 0)      outF[ix] = v;
      else if (epi == 1) outH[ix] = (h16)gelu_exact(v);
      else if (epi == 2) outF[ix] = v + resid[ix];
      else               outH[ix] = (h16)v;
    }
  }
}

// ---------------- softmax over groups of 32 (per head), in-place f32 ----------------
__global__ void softmax_heads(float* __restrict__ q) {
  __shared__ float s[CC];
  int row = blockIdx.x, t = threadIdx.x;
  float v = q[row * CC + t];
  s[t] = v; __syncthreads();
  for (int off = 16; off > 0; off >>= 1) { if ((t & 31) < off) s[t] = fmaxf(s[t], s[t + off]); __syncthreads(); }
  float mx = s[t & ~31];
  __syncthreads();
  float e = expf(v - mx);
  s[t] = e; __syncthreads();
  for (int off = 16; off > 0; off >>= 1) { if ((t & 31) < off) s[t] += s[t + off]; __syncthreads(); }
  float sm = s[t & ~31];
  q[row * CC + t] = e / sm;
}

// ---------------- ksum[b,h,d] = sum_m k[b,m,h,d] ----------------
__global__ void ksum_kernel(const float* __restrict__ k, float* __restrict__ ksum, int Mr) {
  __shared__ float s[256];
  int bh = blockIdx.x; int b = bh >> 3, h = bh & 7;
  int t = threadIdx.x; int d = t & 31, sl = t >> 5;
  float acc = 0.f;
  for (int m = sl; m < Mr; m += 8) acc += k[(size_t)(b * Mr + m) * CC + h * DDIM + d];
  s[t] = acc; __syncthreads();
  if (t < 32) {
    float r = 0.f;
    for (int j = 0; j < 8; ++j) r += s[t + 32 * j];
    ksum[bh * DDIM + t] = r;
  }
}

// ---------------- kv[b,h,d,e] = sum_m k[b,m,h,d]*v[b,m,h,e] ----------------
__global__ void kv_kernel(const float* __restrict__ kg, const float* __restrict__ vg,
                          float* __restrict__ kvg, int Mr) {
  __shared__ float kc[32 * 32];
  __shared__ float vc[32 * 32];
  int bh = blockIdx.x; int b = bh >> 3, h = bh & 7;
  int t = threadIdx.x;
  float acc[4] = {0.f, 0.f, 0.f, 0.f};
  for (int mc = 0; mc < Mr; mc += 32) {
    __syncthreads();
#pragma unroll
    for (int i = 0; i < 4; ++i) {
      int idx = t + i * 256;
      int m = idx >> 5, c = idx & 31;
      kc[idx] = kg[(size_t)(b * Mr + mc + m) * CC + h * DDIM + c];
      vc[idx] = vg[(size_t)(b * Mr + mc + m) * CC + h * DDIM + c];
    }
    __syncthreads();
#pragma unroll
    for (int i = 0; i < 4; ++i) {
      int p = t * 4 + i;
      int d = p >> 5, e = p & 31;
      for (int m = 0; m < 32; ++m) acc[i] += kc[m * 32 + d] * vc[m * 32 + e];
    }
  }
#pragma unroll
  for (int i = 0; i < 4; ++i) kvg[(size_t)bh * 1024 + t * 4 + i] = acc[i];
}

// ---------------- attn combine: out = q@kv / max(q.ksum,1e-6) + q  (f16 out) ----------------
__global__ void attn_combine(const float* __restrict__ qg, const float* __restrict__ ksumg,
                             const float* __restrict__ kvg, h16* __restrict__ outh) {
  __shared__ float kvsh[HH * DDIM * DDIM];  // 32KB
  __shared__ float kssh[CC];
  __shared__ float qsh[CC];
  __shared__ float red[CC];
  int t = threadIdx.x;
  int row0 = blockIdx.x * 8;
  int b = row0 / NN_;
  for (int i = 0; i < 32; ++i) kvsh[t + i * 256] = kvg[(size_t)b * 8192 + t + i * 256];
  kssh[t] = ksumg[b * CC + t];
  __syncthreads();
  int h = t >> 5, e = t & 31;
  for (int r = 0; r < 8; ++r) {
    int row = row0 + r;
    qsh[t] = qg[(size_t)row * CC + t];
    __syncthreads();
    red[t] = qsh[t] * kssh[t];
    __syncthreads();
    for (int off = 16; off > 0; off >>= 1) { if ((t & 31) < off) red[t] += red[t + off]; __syncthreads(); }
    float denom = fmaxf(red[h * 32], 1e-6f);
    float o = 0.f;
#pragma unroll
    for (int d = 0; d < 32; ++d) o += qsh[h * 32 + d] * kvsh[h * 1024 + d * 32 + e];
    outh[(size_t)row * CC + t] = (h16)(o / denom + qsh[t]);
    __syncthreads();
  }
}

// ---------------- gate MLP stage 0: g1 = gelu(coords@W1+b1), K=3 ----------------
__global__ void gate0_kernel(const float* __restrict__ coords, const float* __restrict__ w1,
                             const float* __restrict__ b1, h16* __restrict__ g1) {
  int i = blockIdx.x * blockDim.x + threadIdx.x;
  int row = i >> 9, j = i & 511;
  float c0 = coords[row * 3 + 0], c1 = coords[row * 3 + 1], c2 = coords[row * 3 + 2];
  float v = c0 * w1[j] + c1 * w1[GHN + j] + c2 * w1[2 * GHN + j] + b1[j];
  g1[i] = (h16)gelu_exact(v);
}

// ---------------- gate MLP final: logits = g2@W3+b3 (512->8), softmax ----------------
__global__ void gate_final(const h16* __restrict__ g2, const float* __restrict__ w3,
                           const float* __restrict__ b3, float* __restrict__ gates) {
  __shared__ float s[256];
  __shared__ float lg[8];
  int row = blockIdx.x, t = threadIdx.x;
  int e = t & 7, ks = t >> 3;
  float p = 0.f;
  for (int k = ks; k < GHN; k += 32) p += (float)g2[(size_t)row * GHN + k] * w3[k * 8 + e];
  s[t] = p; __syncthreads();
  if (t < 8) {
    float r = 0.f;
    for (int j = 0; j < 32; ++j) r += s[t + 8 * j];
    lg[t] = r + b3[t];
  }
  __syncthreads();
  if (t < 8) {
    float mx = lg[0];
    for (int j = 1; j < 8; ++j) mx = fmaxf(mx, lg[j]);
    float sm = 0.f;
    for (int j = 0; j < 8; ++j) sm += expf(lg[j] - mx);
    gates[row * 8 + t] = expf(lg[t] - mx) / sm;
  }
}

// ---------------- fused MoE: out = xin + sum_e gates[:,e]*(gelu(xin@W1_e+b1)@W2_e + b2) ----------------
__global__ void __launch_bounds__(256)
moe_kernel(const float* __restrict__ xin, const float* __restrict__ gates,
           const h16* __restrict__ W1, const float* __restrict__ b1,
           const h16* __restrict__ W2, const float* __restrict__ b2,
           float* __restrict__ out) {
  __shared__ h16  Xs[64 * 256];    // 32KB row-major token tile, reused across all experts
  __shared__ h16  W1s[64 * 32];    // 4KB  fragment layout [col][k]
  __shared__ h16  Hs[64 * 64];     // 8KB  row-major hidden chunk (post-GELU)
  __shared__ h16  W2s[256 * 32];   // 16KB fragment layout [col][k]
  __shared__ float Gs[64 * 8];     // 2KB  gates tile
  int t = threadIdx.x;
  int row0 = blockIdx.x * 64;
  int w = t >> 5, lane = t & 31;
  int rt1 = w & 3,  cb1 = (w >> 2) * 32;   // phase-1 tiling: 4x2 over 64x64
  int rt2 = w >> 1, cb2 = (w & 1) * 128;   // phase-2 tiling: 4x2 over 64x256

  // stage X tile (contiguous 64 floats per thread -> vector loads/stores)
  {
    int xrow = t >> 2, xc0 = (t & 3) * 64;
#pragma unroll
    for (int i = 0; i < 64; ++i)
      Xs[xrow * 256 + xc0 + i] = (h16)xin[(size_t)(row0 + xrow) * CC + xc0 + i];
  }
  {
    int gr = t >> 2, gc = (t & 3) * 2;  // 64 rows x 8 gates
    Gs[gr * 8 + gc]     = gates[(size_t)(row0 + gr) * EE + gc];
    Gs[gr * 8 + gc + 1] = gates[(size_t)(row0 + gr) * EE + gc + 1];
  }
  __syncthreads();

  // staging coords
  int w1k = t >> 3, w1c0 = (t & 7) * 8;     // W1 chunk: 32 k x 64 cols
  int w2k = t >> 3, w2c0 = (t & 7) * 32;    // W2 chunk: 32 k x 256 cols

  v8f zero = {};
  v8f oacc[8];
#pragma unroll
  for (int i = 0; i < 8; ++i) oacc[i] = zero;

  for (int e = 0; e < EE; ++e) {
    v8f eacc[8];
#pragma unroll
    for (int i = 0; i < 8; ++i) eacc[i] = zero;

    for (int f = 0; f < HIDN; f += 64) {
      // ---- phase 1: Hs(64x64) = gelu(Xs(64x256) @ W1_e[:, f:f+64] + b1) ----
      v8f h0 = zero, h1 = zero;
      for (int kk = 0; kk < CC; kk += 32) {
        if (kk + 32 < CC)  // prefetch next W1 chunk (global_prefetch_b8)
          __builtin_prefetch(&W1[(size_t)(e * CC + kk + 32 + w1k) * HIDN + f + w1c0], 0, 1);
#pragma unroll
        for (int i = 0; i < 8; ++i)
          W1s[(w1c0 + i) * 32 + w1k] =
              W1[(size_t)(e * CC + kk + w1k) * HIDN + f + w1c0 + i];
        __syncthreads();
        v16h a   = load_a_frag(Xs + rt1 * 16 * CC + kk, CC);
        v16h bb0 = load_b_fragT(W1s + (cb1) * 32);
        v16h bb1 = load_b_fragT(W1s + (cb1 + 16) * 32);
        h0 = wmma16(a, bb0, h0);
        h1 = wmma16(a, bb1, h1);
        __syncthreads();
      }
      {
        int rl = rt1 * 16 + ((lane >> 4) << 3);
        int c0 = cb1 + (lane & 15);
#pragma unroll
        for (int r = 0; r < 8; ++r) {
          Hs[(rl + r) * 64 + c0]      = (h16)gelu_exact(h0[r] + b1[e * HIDN + f + c0]);
          Hs[(rl + r) * 64 + c0 + 16] = (h16)gelu_exact(h1[r] + b1[e * HIDN + f + c0 + 16]);
        }
      }
      __syncthreads();
      // ---- phase 2: eacc(64x256) += Hs(64x64) @ W2_e[f:f+64, :] ----
      for (int kk2 = 0; kk2 < 64; kk2 += 32) {
        __builtin_prefetch(&W2[(size_t)(e * HIDN + f + kk2 + 32 + w2k) * CC + w2c0], 0, 1);
#pragma unroll
        for (int i = 0; i < 32; ++i)
          W2s[(w2c0 + i) * 32 + w2k] =
              W2[(size_t)(e * HIDN + f + kk2 + w2k) * CC + w2c0 + i];
        __syncthreads();
        v16h a2 = load_a_frag(Hs + rt2 * 16 * 64 + kk2, 64);
#pragma unroll
        for (int i = 0; i < 8; ++i) {
          v16h bb = load_b_fragT(W2s + (cb2 + i * 16) * 32);
          eacc[i] = wmma16(a2, bb, eacc[i]);
        }
        __syncthreads();
      }
    }
    // gate + bias accumulation into output accumulator
    {
      int rl = rt2 * 16 + ((lane >> 4) << 3);
#pragma unroll
      for (int i = 0; i < 8; ++i) {
        int col = cb2 + i * 16 + (lane & 15);
        float bb2 = b2[e * CC + col];
#pragma unroll
        for (int r = 0; r < 8; ++r) {
          float g = Gs[(rl + r) * EE + e];
          oacc[i][r] += g * (eacc[i][r] + bb2);
        }
      }
    }
  }
  // residual add + store
  {
    int rl = rt2 * 16 + ((lane >> 4) << 3);
#pragma unroll
    for (int i = 0; i < 8; ++i) {
      int col = cb2 + i * 16 + (lane & 15);
#pragma unroll
      for (int r = 0; r < 8; ++r) {
        size_t idx = (size_t)(row0 + rl + r) * CC + col;
        out[idx] = xin[idx] + oacc[i][r];
      }
    }
  }
}

// =====================================================================
extern "C" void kernel_launch(void* const* d_in, const int* in_sizes, int n_in,
                              void* d_out, int out_size, void* d_ws, size_t ws_size,
                              hipStream_t stream) {
  (void)in_sizes; (void)n_in; (void)out_size; (void)ws_size;
  const float* x          = (const float*)d_in[0];
  const float* branch     = (const float*)d_in[1];
  const float* coords     = (const float*)d_in[2];
  const float* ln_cross_g = (const float*)d_in[3];
  const float* ln_cross_b = (const float*)d_in[4];
  const float* ln_branch_g= (const float*)d_in[5];
  const float* ln_branch_b= (const float*)d_in[6];
  const float* ln_self_g  = (const float*)d_in[7];
  const float* ln_self_b  = (const float*)d_in[8];
  const float* ca_wq = (const float*)d_in[9];  const float* ca_bq = (const float*)d_in[10];
  const float* ca_wk = (const float*)d_in[11]; const float* ca_bk = (const float*)d_in[12];
  const float* ca_wv = (const float*)d_in[13]; const float* ca_bv = (const float*)d_in[14];
  const float* ca_wp = (const float*)d_in[15]; const float* ca_bp = (const float*)d_in[16];
  const float* sa_wq = (const float*)d_in[17]; const float* sa_bq = (const float*)d_in[18];
  const float* sa_wk = (const float*)d_in[19]; const float* sa_bk = (const float*)d_in[20];
  const float* sa_wv = (const float*)d_in[21]; const float* sa_bv = (const float*)d_in[22];
  const float* sa_wp = (const float*)d_in[23]; const float* sa_bp = (const float*)d_in[24];
  const float* gate_w1 = (const float*)d_in[25]; const float* gate_b1 = (const float*)d_in[26];
  const float* gate_w2 = (const float*)d_in[27]; const float* gate_b2 = (const float*)d_in[28];
  const float* gate_w3 = (const float*)d_in[29]; const float* gate_b3 = (const float*)d_in[30];
  const float* m1_w1 = (const float*)d_in[31]; const float* m1_b1 = (const float*)d_in[32];
  const float* m1_w2 = (const float*)d_in[33]; const float* m1_b2 = (const float*)d_in[34];
  const float* m2_w1 = (const float*)d_in[35]; const float* m2_b1 = (const float*)d_in[36];
  const float* m2_w2 = (const float*)d_in[37]; const float* m2_b2 = (const float*)d_in[38];
  float* outp = (float*)d_out;

  const int RN = BB_ * NN_;  // 8192 query rows
  const int RM = BB_ * MM_;  // 4096 branch rows

  // ---- carve workspace ----
  char* p = (char*)d_ws;
  auto carve = [&](size_t bytes) -> void* {
    void* r = (void*)p;
    p += (bytes + 255) & ~(size_t)255;
    return r;
  };
  float* x1    = (float*)carve((size_t)RN * CC * 4);
  float* x2    = (float*)carve((size_t)RN * CC * 4);
  h16*   lnx   = (h16*)  carve((size_t)RN * CC * 2);
  h16*   lny   = (h16*)  carve((size_t)RM * CC * 2);
  float* qb    = (float*)carve((size_t)RN * CC * 4);
  float* kb    = (float*)carve((size_t)RN * CC * 4);
  float* vb    = (float*)carve((size_t)RN * CC * 4);
  float* ksumb = (float*)carve((size_t)BB_ * HH * DDIM * 4);
  float* kvb   = (float*)carve((size_t)BB_ * HH * DDIM * DDIM * 4);
  h16*   attnb = (h16*)  carve((size_t)RN * CC * 2);
  h16*   g1b   = (h16*)  carve((size_t)RN * GHN * 2);
  h16*   g2b   = (h16*)  carve((size_t)RN * GHN * 2);
  float* gatesb= (float*)carve((size_t)RN * EE * 4);
  h16* caq_h = (h16*)carve((size_t)CC*CC*2);  h16* cak_h = (h16*)carve((size_t)CC*CC*2);
  h16* cav_h = (h16*)carve((size_t)CC*CC*2);  h16* cap_h = (h16*)carve((size_t)CC*CC*2);
  h16* saq_h = (h16*)carve((size_t)CC*CC*2);  h16* sak_h = (h16*)carve((size_t)CC*CC*2);
  h16* sav_h = (h16*)carve((size_t)CC*CC*2);  h16* sap_h = (h16*)carve((size_t)CC*CC*2);
  h16* gw2_h = (h16*)carve((size_t)GHN*GHN*2);
  h16* m1w1_h = (h16*)carve((size_t)EE*CC*HIDN*2);
  h16* m1w2_h = (h16*)carve((size_t)EE*HIDN*CC*2);
  h16* m2w1_h = (h16*)carve((size_t)EE*CC*HIDN*2);
  h16* m2w2_h = (h16*)carve((size_t)EE*HIDN*CC*2);

  auto cvt = [&](const float* s, h16* d, int n) {
    cvt_f32_f16<<<(n + 255) / 256, 256, 0, stream>>>(s, d, n);
  };
  cvt(ca_wq, caq_h, CC*CC);  cvt(ca_wk, cak_h, CC*CC);
  cvt(ca_wv, cav_h, CC*CC);  cvt(ca_wp, cap_h, CC*CC);
  cvt(sa_wq, saq_h, CC*CC);  cvt(sa_wk, sak_h, CC*CC);
  cvt(sa_wv, sav_h, CC*CC);  cvt(sa_wp, sap_h, CC*CC);
  cvt(gate_w2, gw2_h, GHN*GHN);
  cvt(m1_w1, m1w1_h, EE*CC*HIDN);  cvt(m1_w2, m1w2_h, EE*HIDN*CC);
  cvt(m2_w1, m2w1_h, EE*CC*HIDN);  cvt(m2_w2, m2w2_h, EE*HIDN*CC);

  auto gemm = [&](const h16* A, const h16* B, const float* bias, const float* resid,
                  float* oF, h16* oH, int M, int N, int K, int epi) {
    dim3 grid(N / 128, M / 64);
    gemm_kernel<<<grid, 256, 0, stream>>>(A, B, bias, resid, oF, oH, M, N, K, epi);
  };

  // ---- gate MLP (shared) ----
  gate0_kernel<<<(RN * GHN) / 256, 256, 0, stream>>>(coords, gate_w1, gate_b1, g1b);
  gemm(g1b, gw2_h, gate_b2, nullptr, nullptr, g2b, RN, GHN, GHN, /*gelu->f16*/1);
  gate_final<<<RN, 256, 0, stream>>>(g2b, gate_w3, gate_b3, gatesb);

  // ---- cross attention ----
  ln_kernel<<<RN, CC, 0, stream>>>(x, ln_cross_g, ln_cross_b, lnx);
  ln_kernel<<<RM, CC, 0, stream>>>(branch, ln_branch_g, ln_branch_b, lny);
  gemm(lnx, caq_h, ca_bq, nullptr, qb, nullptr, RN, CC, CC, 0);
  gemm(lny, cak_h, ca_bk, nullptr, kb, nullptr, RM, CC, CC, 0);
  gemm(lny, cav_h, ca_bv, nullptr, vb, nullptr, RM, CC, CC, 0);
  softmax_heads<<<RN, CC, 0, stream>>>(qb);
  softmax_heads<<<RM, CC, 0, stream>>>(kb);
  ksum_kernel<<<BB_ * HH, 256, 0, stream>>>(kb, ksumb, MM_);
  kv_kernel<<<BB_ * HH, 256, 0, stream>>>(kb, vb, kvb, MM_);
  attn_combine<<<RN / 8, 256, 0, stream>>>(qb, ksumb, kvb, attnb);
  gemm(attnb, cap_h, ca_bp, x, x1, nullptr, RN, CC, CC, /*+resid*/2);

  // ---- MoE 1: x2 = x1 + moe(x1) ----
  moe_kernel<<<RN / 64, 256, 0, stream>>>(x1, gatesb, m1w1_h, m1_b1, m1w2_h, m1_b2, x2);

  // ---- self attention ----
  ln_kernel<<<RN, CC, 0, stream>>>(x2, ln_self_g, ln_self_b, lnx);
  gemm(lnx, saq_h, sa_bq, nullptr, qb, nullptr, RN, CC, CC, 0);
  gemm(lnx, sak_h, sa_bk, nullptr, kb, nullptr, RN, CC, CC, 0);
  gemm(lnx, sav_h, sa_bv, nullptr, vb, nullptr, RN, CC, CC, 0);
  softmax_heads<<<RN, CC, 0, stream>>>(qb);
  softmax_heads<<<RN, CC, 0, stream>>>(kb);
  ksum_kernel<<<BB_ * HH, 256, 0, stream>>>(kb, ksumb, NN_);
  kv_kernel<<<BB_ * HH, 256, 0, stream>>>(kb, vb, kvb, NN_);
  attn_combine<<<RN / 8, 256, 0, stream>>>(qb, ksumb, kvb, attnb);
  gemm(attnb, sap_h, sa_bp, x2, x1, nullptr, RN, CC, CC, /*+resid*/2);

  // ---- MoE 2: out = x1 + moe(x1) ----
  moe_kernel<<<RN / 64, 256, 0, stream>>>(x1, gatesb, m2w1_h, m2_b1, m2w2_h, m2_b2, outp);
}
